// Mambalayer_16475494547742
// MI455X (gfx1250) — matrix operand order repaired and checked by reading, not compiled, not run on hardware
//
#include <hip/hip_runtime.h>
#include <hip/hip_bf16.h>

// ---------------------------------------------------------------------------
// Mamba layer for MI455X (gfx1250, wave32, WMMA + Tensor Data Mover).
// GEMMs: v_wmma_f32_16x16x32_f16, 128x128x32 workgroup tiles, 8 waves,
//        each wave computes a 32x64 sub-tile (2x4 WMMA accumulators).
//        A-tile is DMA'd into LDS by the TDM (tensor_load_to_lds, TENSORcnt),
//        using TDM row padding to produce the 96B-stride LDS layout directly.
// Scan:  persistent per-channel VALU loop (sequential in L).
// ---------------------------------------------------------------------------

typedef __attribute__((ext_vector_type(16))) _Float16     v16h;
typedef __attribute__((ext_vector_type(8)))  _Float16     v8h;
typedef __attribute__((ext_vector_type(8)))  float        v8f;
typedef __attribute__((ext_vector_type(4)))  unsigned int v4u;
typedef __attribute__((ext_vector_type(8)))  unsigned int v8u;

#define BM 128
#define BN 128
#define BK 32
#define AROW 48          // As row stride in halves: 64B data + 32B TDM pad = 96B
#define BROW 144         // Bs row stride in halves: 288B (16B multiple)
#define ABYTES (BM * AROW * 2)          // 12288
#define SMEM_BYTES (ABYTES + BK * BROW * 2)  // 12288 + 9216 = 21504

// C[M,N] = A[M,K] * W[N,K]^T   (A,W f16 row-major, C f32, ldc >= N)
__global__ __launch_bounds__(256) void wmma_gemm_nt(
    const _Float16* __restrict__ A, const _Float16* __restrict__ W,
    float* __restrict__ C, int M, int N, int K, int ldc)
{
    extern __shared__ __align__(16) char smem[];
    _Float16* As = (_Float16*)smem;              // [BM][AROW], LDS offset 0
    _Float16* Bs = (_Float16*)(smem + ABYTES);   // [BK][BROW]

    const int tid  = threadIdx.x;
    const int lane = tid & 31;
    const int wave = tid >> 5;
    const int wm   = wave >> 1;   // 0..3 : 32-row strip
    const int wn   = wave & 1;    // 0..1 : 64-col strip
    const int m0   = blockIdx.y * BM;
    const int n0   = blockIdx.x * BN;

    const int r  = lane & 15;     // A-frag row / C-frag col
    const int hi = lane >> 4;     // lane half select

    v8f acc[2][4];
    const v8f z8 = {0.f,0.f,0.f,0.f,0.f,0.f,0.f,0.f};
    #pragma unroll
    for (int i = 0; i < 2; ++i)
        #pragma unroll
        for (int j = 0; j < 4; ++j) acc[i][j] = z8;

    for (int k0 = 0; k0 < K; k0 += BK) {
        // ---- A tile via Tensor Data Mover: 128 rows x 32 halves -> As ----
        // TDM pads each 64B row with 32B -> 96B LDS row stride (= AROW halves).
        if (wave == 0) {
            unsigned long long ga =
                (unsigned long long)(const void*)(A + (size_t)m0 * K + k0);
            v4u g0;
            g0[0] = 1u;                                   // count=1, user D#
            g0[1] = 0u;                                   // lds_addr = 0 (As base)
            g0[2] = (unsigned int)ga;                     // global_addr[31:0]
            g0[3] = ((unsigned int)(ga >> 32) & 0x01FFFFFFu) | (2u << 30); // addr hi | type=2
            v8u g1;
            g1[0] = (1u << 16)      // data_size = 2 bytes
                  | (1u << 20)      // pad_enable
                  | (3u << 22)      // pad_interval: 16 dwords (one 64B row)
                  | (7u << 25);     // pad_amount: 8 dwords (32B)
            g1[1] = ((unsigned int)K & 0xFFFFu) << 16;                  // tensor_dim0 lo
            g1[2] = ((unsigned int)K >> 16) |
                    (((unsigned int)M & 0xFFFFu) << 16);                // dim0 hi | tensor_dim1 lo
            g1[3] = ((unsigned int)M >> 16) | ((unsigned int)BK << 16); // dim1 hi | tile_dim0=32
            g1[4] = (unsigned int)BM;                                   // tile_dim1=128 (tile_dim2=0)
            g1[5] = (unsigned int)K;                                    // tensor_dim0_stride lo32
            g1[6] = 0u;
            g1[7] = 0u;
            asm volatile("tensor_load_to_lds %0, %1"
                         :: "s"(g0), "s"(g1) : "memory");
        }

        // ---- stage W tile transposed : Bs[k][n]  (all 8 waves) ----
        #pragma unroll
        for (int it = 0; it < 2; ++it) {
            int idx  = tid + it * 256;
            int nrow = idx >> 2;               // 0..127
            int seg  = (idx & 3) * 8;          // k sub-offset
            int gn   = n0 + nrow;
            v8h wv;
            if (gn < N) {
                wv = *(const v8h*)(W + (size_t)gn * K + k0 + seg);
            } else {
                const _Float16 h0 = (_Float16)0.f;
                wv = (v8h){h0,h0,h0,h0,h0,h0,h0,h0};
            }
            #pragma unroll
            for (int q = 0; q < 8; ++q) Bs[(seg + q) * BROW + nrow] = wv[q];
        }
        // prefetch next W K-tile into cache while we compute
        if (k0 + BK < K && n0 + (tid >> 1) < N)
            __builtin_prefetch(W + (size_t)(n0 + (tid >> 1)) * K + (k0 + BK), 0, 1);

        if (wave == 0)
            __builtin_amdgcn_s_wait_tensorcnt(0);   // TDM writes to As complete
        __syncthreads();

        // ---- load fragments ----
        // A frag (16x32 f16): lane<16 -> row=r, K {kh..kh+7, kh+16..kh+23}, kh=hi*8
        v16h af[2];
        #pragma unroll
        for (int i = 0; i < 2; ++i) {
            const _Float16* ap = As + (wm * 32 + i * 16 + r) * AROW + hi * 8;
            v8h lo  = *(const v8h*)ap;
            v8h hi8 = *(const v8h*)(ap + 16);
            af[i] = __builtin_shufflevector(lo, hi8,
                0,1,2,3,4,5,6,7,8,9,10,11,12,13,14,15);
        }
        // B frag (32x16 f16): lane -> col=r, K {kh..kh+7, kh+16..kh+23}, kh=hi*8
        v16h bf[4];
        #pragma unroll
        for (int j = 0; j < 4; ++j) {
            const _Float16* bp = Bs + (hi * 8) * BROW + wn * 64 + j * 16 + r;
            #pragma unroll
            for (int s = 0; s < 8; ++s) {
                bf[j][s]     = bp[s * BROW];
                bf[j][8 + s] = bp[(16 + s) * BROW];
            }
        }

        #pragma unroll
        for (int i = 0; i < 2; ++i)
            #pragma unroll
            for (int j = 0; j < 4; ++j)
                acc[i][j] = __builtin_amdgcn_wmma_f32_16x16x32_f16(
                    false, af[i], false, bf[j], (short)0, acc[i][j], false, false);

        __syncthreads();
    }

    // ---- epilogue: C frag layout (lane<16: M=rr,N=lane ; lane>=16: M=8+rr) ----
    const int nn  = lane & 15;
    const int mof = hi * 8;
    #pragma unroll
    for (int i = 0; i < 2; ++i) {
        #pragma unroll
        for (int j = 0; j < 4; ++j) {
            int nidx = n0 + wn * 64 + j * 16 + nn;
            if (nidx < N) {
                float* cp = C + (size_t)(m0 + wm * 32 + i * 16 + mof) * ldc + nidx;
                #pragma unroll
                for (int rr = 0; rr < 8; ++rr)
                    cp[(size_t)rr * ldc] = acc[i][j][rr];
            }
        }
    }
}

// ---------------------------------------------------------------------------
// Elementwise / helper kernels
// ---------------------------------------------------------------------------
__global__ void f32_to_f16_k(const float* __restrict__ s, _Float16* __restrict__ d, int n) {
    for (int i = blockIdx.x * blockDim.x + threadIdx.x; i < n; i += gridDim.x * blockDim.x)
        d[i] = (_Float16)s[i];
}

// x_proj_w [48,1024] -> padded f16 [64,1024]
__global__ void pad_xp_w_k(const float* __restrict__ w, _Float16* __restrict__ d) {
    int n = 64 * 1024;
    for (int i = blockIdx.x * blockDim.x + threadIdx.x; i < n; i += gridDim.x * blockDim.x) {
        int row = i >> 10, col = i & 1023;
        d[i] = (row < 48) ? (_Float16)w[row * 1024 + col] : (_Float16)0.f;
    }
}

// dt_proj_w [1024,16] -> padded f16 [1024,32]
__global__ void pad_dt_w_k(const float* __restrict__ w, _Float16* __restrict__ d) {
    int n = 1024 * 32;
    for (int i = blockIdx.x * blockDim.x + threadIdx.x; i < n; i += gridDim.x * blockDim.x) {
        int row = i >> 5, col = i & 31;
        d[i] = (col < 16) ? (_Float16)w[row * 16 + col] : (_Float16)0.f;
    }
}

// x_dbl [M,64] cols 0..15 -> dth f16 [M,32] (K padded)
__global__ void pack_dth_k(const float* __restrict__ xdbl, _Float16* __restrict__ d, int M) {
    int n = M * 32;
    for (int i = blockIdx.x * blockDim.x + threadIdx.x; i < n; i += gridDim.x * blockDim.x) {
        int row = i >> 5, col = i & 31;
        d[i] = (col < 16) ? (_Float16)xdbl[row * 64 + col] : (_Float16)0.f;
    }
}

// depthwise causal conv over time + bias + SiLU.  xz: [M, 2048] (xs = cols 0..1023)
__global__ void conv_silu_k(const float* __restrict__ xz,
                            const float* __restrict__ cw, const float* __restrict__ cb,
                            float* __restrict__ u, _Float16* __restrict__ uh,
                            int M, int L) {
    int n = M * 1024;
    for (int i = blockIdx.x * blockDim.x + threadIdx.x; i < n; i += gridDim.x * blockDim.x) {
        int row = i >> 10;           // b*L + l
        int d   = i & 1023;
        int l   = row & (L - 1);
        float acc = cb[d];
        #pragma unroll
        for (int t = 0; t < 4; ++t) {
            int lt = l - 3 + t;
            if (lt >= 0)
                acc += xz[(size_t)(row - 3 + t) * 2048 + d] * cw[d * 4 + t];
        }
        float v = acc / (1.f + __expf(-acc));   // SiLU
        u[i]  = v;
        uh[i] = (_Float16)v;
    }
}

// dt = softplus(dt_raw + bias)  (in place)
__global__ void softplus_bias_k(float* __restrict__ dt, const float* __restrict__ b, int n) {
    for (int i = blockIdx.x * blockDim.x + threadIdx.x; i < n; i += gridDim.x * blockDim.x) {
        int d = i & 1023;
        float v = dt[i] + b[d];
        dt[i] = (v > 20.f) ? v : log1pf(__expf(v));
    }
}

// selective scan + skip + gate: one thread per (b,d) channel, serial over L.
__global__ __launch_bounds__(256) void scan_gate_k(
    const float* __restrict__ dt, const float* __restrict__ u,
    const float* __restrict__ xdbl, const float* __restrict__ xz,
    const float* __restrict__ A_log, const float* __restrict__ Dp,
    _Float16* __restrict__ gh, int Bsz, int L) {
    int idx = blockIdx.x * blockDim.x + threadIdx.x;
    if (idx >= Bsz * 1024) return;
    int b = idx >> 10, d = idx & 1023;
    float Ar[16];
    #pragma unroll
    for (int s = 0; s < 16; ++s) Ar[s] = -__expf(A_log[d * 16 + s]);
    float h[16];
    #pragma unroll
    for (int s = 0; s < 16; ++s) h[s] = 0.f;
    const float dp = Dp[d];
    for (int l = 0; l < L; ++l) {
        size_t row = (size_t)(b * L + l);
        float dtv = dt[row * 1024 + d];
        float uv  = u[row * 1024 + d];
        const float* bc = xdbl + row * 64;
        float dtu = dtv * uv;
        float acc = 0.f;
        #pragma unroll
        for (int s = 0; s < 16; ++s) {
            float dA = __expf(dtv * Ar[s]);
            h[s] = dA * h[s] + dtu * bc[16 + s];
            acc += h[s] * bc[32 + s];
        }
        float yv = acc + dp * uv;
        float zv = xz[row * 2048 + 1024 + d];
        float g  = yv * (zv / (1.f + __expf(-zv)));   // y * silu(z)
        gh[row * 1024 + d] = (_Float16)g;
    }
}

// out = LN(a + b [+ addb]) * g + be ; optional f16 copy. One block per row, D=256.
__global__ __launch_bounds__(256) void add_ln_k(
    const float* __restrict__ a, const float* __restrict__ b,
    const float* __restrict__ addb,
    const float* __restrict__ g, const float* __restrict__ be,
    float* __restrict__ o, _Float16* __restrict__ oh) {
    __shared__ float red[256];
    const int row = blockIdx.x, t = threadIdx.x, D = 256;
    size_t base = (size_t)row * D + t;
    float v = a[base] + b[base] + (addb ? addb[t] : 0.f);
    red[t] = v; __syncthreads();
    for (int s = 128; s > 0; s >>= 1) {
        if (t < s) red[t] += red[t + s];
        __syncthreads();
    }
    float mean = red[0] / (float)D; __syncthreads();
    float c = v - mean;
    red[t] = c * c; __syncthreads();
    for (int s = 128; s > 0; s >>= 1) {
        if (t < s) red[t] += red[t + s];
        __syncthreads();
    }
    float var = red[0] / (float)D;
    float rres = c * rsqrtf(var + 1e-6f) * g[t] + be[t];
    o[base] = rres;
    if (oh) oh[base] = (_Float16)rres;
}

// hfh = f16(relu(hf + b1))
__global__ void relu_bias_k(const float* __restrict__ hf, const float* __restrict__ b1,
                            _Float16* __restrict__ hfh, int n) {
    for (int i = blockIdx.x * blockDim.x + threadIdx.x; i < n; i += gridDim.x * blockDim.x) {
        float v = hf[i] + b1[i & 1023];
        hfh[i] = (_Float16)(v > 0.f ? v : 0.f);
    }
}

// ---------------------------------------------------------------------------
extern "C" void kernel_launch(void* const* d_in, const int* in_sizes, int n_in,
                              void* d_out, int out_size, void* d_ws, size_t ws_size,
                              hipStream_t stream) {
    const float* x      = (const float*)d_in[0];
    const float* w_in   = (const float*)d_in[1];
    const float* conv_w = (const float*)d_in[2];
    const float* conv_b = (const float*)d_in[3];
    const float* w_xp   = (const float*)d_in[4];
    const float* w_dt   = (const float*)d_in[5];
    const float* b_dt   = (const float*)d_in[6];
    const float* A_log  = (const float*)d_in[7];
    const float* Dp     = (const float*)d_in[8];
    const float* w_out  = (const float*)d_in[9];
    const float* ln1g   = (const float*)d_in[10];
    const float* ln1b   = (const float*)d_in[11];
    const float* ln2g   = (const float*)d_in[12];
    const float* ln2b   = (const float*)d_in[13];
    const float* w_f1   = (const float*)d_in[14];
    const float* b_f1   = (const float*)d_in[15];
    const float* w_f2   = (const float*)d_in[16];
    const float* b_f2   = (const float*)d_in[17];
    float* out = (float*)d_out;

    const int Bsz = 2, L = 2048, D = 256, DI = 1024;
    const int M = Bsz * L;   // 4096

    // bump allocator over d_ws
    char* p = (char*)d_ws;
    auto alloc = [&](size_t bytes) -> char* {
        char* r = p; p += (bytes + 255) & ~(size_t)255; return r;
    };
    _Float16* xh     = (_Float16*)alloc((size_t)M * D * 2);
    _Float16* wh_in  = (_Float16*)alloc((size_t)2048 * 256 * 2);
    _Float16* uh     = (_Float16*)alloc((size_t)M * DI * 2);
    _Float16* wh_xp  = (_Float16*)alloc((size_t)64 * 1024 * 2);
    _Float16* dth    = (_Float16*)alloc((size_t)M * 32 * 2);
    _Float16* wh_dt  = (_Float16*)alloc((size_t)1024 * 32 * 2);
    _Float16* gh     = (_Float16*)alloc((size_t)M * DI * 2);
    _Float16* wh_out = (_Float16*)alloc((size_t)256 * 1024 * 2);
    _Float16* o1h    = (_Float16*)alloc((size_t)M * D * 2);
    _Float16* wh_f1  = (_Float16*)alloc((size_t)1024 * 256 * 2);
    _Float16* hfh    = (_Float16*)alloc((size_t)M * DI * 2);
    _Float16* wh_f2  = (_Float16*)alloc((size_t)256 * 1024 * 2);

    float* xz   = (float*)alloc((size_t)M * 2048 * 4);
    float* u    = (float*)alloc((size_t)M * DI * 4);
    float* xdbl = (float*)alloc((size_t)M * 64 * 4);
    float* dtf  = (float*)alloc((size_t)M * DI * 4);
    float* mo   = (float*)alloc((size_t)M * D * 4);
    float* out1 = (float*)alloc((size_t)M * D * 4);
    float* hf   = (float*)alloc((size_t)M * DI * 4);
    float* f2   = (float*)alloc((size_t)M * D * 4);

    const dim3 blk(256);
    auto gs = [](int n) { return dim3((unsigned)((n + 255) / 256)); };

    // weight / input conversions
    f32_to_f16_k<<<gs(M * D), blk, 0, stream>>>(x, xh, M * D);
    f32_to_f16_k<<<gs(2048 * 256), blk, 0, stream>>>(w_in, wh_in, 2048 * 256);
    f32_to_f16_k<<<gs(256 * 1024), blk, 0, stream>>>(w_out, wh_out, 256 * 1024);
    f32_to_f16_k<<<gs(1024 * 256), blk, 0, stream>>>(w_f1, wh_f1, 1024 * 256);
    f32_to_f16_k<<<gs(256 * 1024), blk, 0, stream>>>(w_f2, wh_f2, 256 * 1024);
    pad_xp_w_k<<<gs(64 * 1024), blk, 0, stream>>>(w_xp, wh_xp);
    pad_dt_w_k<<<gs(1024 * 32), blk, 0, stream>>>(w_dt, wh_dt);

    // in_proj: xz[M,2048] = x[M,256] @ in_proj_w^T
    wmma_gemm_nt<<<dim3(2048 / BN, M / BM), blk, SMEM_BYTES, stream>>>(
        xh, wh_in, xz, M, 2048, 256, 2048);

    // depthwise conv + SiLU -> u, uh
    conv_silu_k<<<gs(M * DI), blk, 0, stream>>>(xz, conv_w, conv_b, u, uh, M, L);

    // x_proj: xdbl[M,48] (ld 64) = u @ x_proj_w^T
    wmma_gemm_nt<<<dim3(1, M / BM), blk, SMEM_BYTES, stream>>>(
        uh, wh_xp, xdbl, M, 48, 1024, 64);

    // dt_proj (K padded 16->32)
    pack_dth_k<<<gs(M * 32), blk, 0, stream>>>(xdbl, dth, M);
    wmma_gemm_nt<<<dim3(1024 / BN, M / BM), blk, SMEM_BYTES, stream>>>(
        dth, wh_dt, dtf, M, 1024, 32, 1024);
    softplus_bias_k<<<gs(M * DI), blk, 0, stream>>>(dtf, b_dt, M * DI);

    // selective scan + D-skip + SiLU gate -> gh (f16)
    scan_gate_k<<<gs(Bsz * DI), blk, 0, stream>>>(
        dtf, u, xdbl, xz, A_log, Dp, gh, Bsz, L);

    // out_proj: mo[M,256] = gh @ out_proj_w^T
    wmma_gemm_nt<<<dim3(256 / BN, M / BM), blk, SMEM_BYTES, stream>>>(
        gh, wh_out, mo, M, 256, 1024, 256);

    // out1 = LN(x + mamba_out)
    add_ln_k<<<dim3(M), blk, 0, stream>>>(x, mo, nullptr, ln1g, ln1b, out1, o1h);

    // ffn1 + relu
    wmma_gemm_nt<<<dim3(1024 / BN, M / BM), blk, SMEM_BYTES, stream>>>(
        o1h, wh_f1, hf, M, 1024, 256, 1024);
    relu_bias_k<<<gs(M * DI), blk, 0, stream>>>(hf, b_f1, hfh, M * DI);

    // ffn2
    wmma_gemm_nt<<<dim3(256 / BN, M / BM), blk, SMEM_BYTES, stream>>>(
        hfh, wh_f2, f2, M, 256, 1024, 256);

    // final: out = LN(out1 + f + b2)
    add_ln_k<<<dim3(M), blk, 0, stream>>>(out1, f2, b_f2, ln2g, ln2b, out, nullptr);
}
